// TimeWiseMultiheadAttention_53944789237984
// MI455X (gfx1250) — matrix operand order repaired and checked by reading, not compiled, not run on hardware
//
#include <hip/hip_runtime.h>
#include <hip/hip_bf16.h>

typedef __attribute__((ext_vector_type(16))) _Float16 v16h;
typedef __attribute__((ext_vector_type(8)))  _Float16 v8h;
typedef __attribute__((ext_vector_type(8)))  float    v8f;
typedef __attribute__((ext_vector_type(4)))  unsigned int v4u;
typedef __attribute__((ext_vector_type(8)))  int v8i;
typedef __attribute__((ext_vector_type(4)))  int v4i;

#define SS 512
#define EE 1024
#define HH 16
#define DD 64
#define BV 32          // B*V
#define BHN 512        // BV*HH
#define MROWS 16384    // BV*SS
#define NQKV 3072

// ---------------- WMMA fragment loaders (wave32, gfx1250 layouts) ----------------
// A (16x32 f16, MxK): lane L holds row M=L%16; lanes<16 hold K{0..15},
// lanes>=16 hold K{16..31}; 2 halfs per VGPR -> two contiguous 8-half loads.
__device__ __forceinline__ v16h load_a16(const _Float16* __restrict__ base, int ld) {
    const int lane = threadIdx.x & 31;
    const int row  = lane & 15;
    const int kb   = (lane >> 4) << 4;      // 0 or 16
    const _Float16* p = base + (size_t)row * ld + kb;
    v8h lo = *(const v8h*)(p);              // K = kb .. kb+7
    v8h hi = *(const v8h*)(p + 8);          // K = kb+8 .. kb+15
    v16h r;
#pragma unroll
    for (int i = 0; i < 8; ++i) { r[i] = lo[i]; r[i + 8] = hi[i]; }
    return r;
}

// B (32x16 f16, KxN) loaded from a transposed (N,K) row-major array:
// lane L holds column N=L%16; lanes<16 hold K=0..15, lanes>=16 hold K=16..31
// (contiguous k order, same as load_a16) -> one contiguous 16-half (32B) load per lane.
__device__ __forceinline__ v16h load_bT16(const _Float16* __restrict__ baseT, int ld) {
    const int lane = threadIdx.x & 31;
    const int n  = lane & 15;
    const int kb = (lane >> 4) << 4;        // 0 or 16
    return *(const v16h*)(baseT + (size_t)n * ld + kb);
}

// ---------------- Tensor Data Mover: 2D tile (global -> LDS), data_size = 4B ----------------
// row_dw: row length in DWORDs; nrows: number of rows; stride_dw: row stride in DWORDs.
// Tile rows are packed contiguously into LDS at lds_off.
__device__ __forceinline__ void tdm_load_2d(unsigned lds_off, const void* gaddr,
                                            unsigned row_dw, unsigned nrows,
                                            unsigned stride_dw) {
    unsigned long long ga = (unsigned long long)gaddr;
    v4u g0;
    g0[0] = 1u;                                           // count=1, user D#
    g0[1] = lds_off;                                      // lds_addr (bytes)
    g0[2] = (unsigned)(ga & 0xFFFFFFFFu);                 // global_addr[31:0]
    g0[3] = (unsigned)((ga >> 32) & 0x1FFFFFFu)           // global_addr[56:32]
            | (2u << 30);                                 // type = 2 ("image")
    v8i g1;
    g1[0] = (int)(2u << 16);                              // data_size = 2 (4 bytes)
    g1[1] = (int)((row_dw & 0xFFFFu) << 16);              // tensor_dim0[15:0]
    g1[2] = (int)((row_dw >> 16) | ((nrows & 0xFFFFu) << 16));  // dim0 hi | tensor_dim1 lo
    g1[3] = (int)((nrows >> 16) | ((row_dw & 0xFFFFu) << 16));  // dim1 hi | tile_dim0
    g1[4] = (int)(nrows & 0xFFFFu);                       // tile_dim1 (tile_dim2 = 0)
    g1[5] = (int)stride_dw;                               // tensor_dim0_stride[31:0]
    g1[6] = 0;                                            // stride hi | dim1_stride lo
    g1[7] = 0;
    v4i g2 = {};                                          // 2D tile: groups 2/3 unused
    v4i g3 = {};
    v8i g4 = {};                                          // 6-arg toolchain form: extra group
    __builtin_amdgcn_tensor_load_to_lds(g0, g1, g2, g3, g4, 0);
}

// ---------------- conversion kernels ----------------
__global__ void cvt_x_kernel(const float* __restrict__ x, _Float16* __restrict__ xh, int n) {
    int i = blockIdx.x * blockDim.x + threadIdx.x;
    if (i < n) xh[i] = (_Float16)x[i];
}

// w: (K,N) row-major fp32  ->  wT: (N,K) row-major f16
__global__ void cvt_wT_kernel(const float* __restrict__ w, _Float16* __restrict__ wT,
                              int K, int N) {
    int i = blockIdx.x * blockDim.x + threadIdx.x;
    if (i >= K * N) return;
    int n = i / K, k = i - n * K;
    wT[i] = (_Float16)w[(size_t)k * N + n];
}

// ---------------- QKV projection GEMM: Xh(16384x1024) @ Wqkv + b ----------------
__global__ __launch_bounds__(256)
void qkv_gemm_kernel(const _Float16* __restrict__ Xh,
                     const _Float16* __restrict__ WT,    // (3072, 1024)
                     const float*    __restrict__ bias,  // (3072)
                     _Float16* __restrict__ Qh,          // (512, 512, 64)
                     _Float16* __restrict__ Kh,          // (512, 512, 64)
                     _Float16* __restrict__ Vt)          // (512, 64, 512)
{
    const int lane   = threadIdx.x & 31;
    const int wave   = blockIdx.x * (blockDim.x >> 5) + (threadIdx.x >> 5);
    const int ntiles = NQKV / 64;           // 48
    const int mt = wave / ntiles;
    const int nt = wave - mt * ntiles;
    if (mt >= MROWS / 16) return;
    const int m0 = mt * 16, n0 = nt * 64;

    v8f acc[4] = {};
    const _Float16* Arow = Xh + (size_t)m0 * EE;
    for (int k0 = 0; k0 < EE; k0 += 32) {
        v16h a = load_a16(Arow + k0, EE);
        __builtin_prefetch(Arow + k0 + 32, 0, 1);   // global_prefetch_b8
#pragma unroll
        for (int j = 0; j < 4; ++j) {
            v16h b = load_bT16(WT + (size_t)(n0 + 16 * j) * EE + k0, EE);
            acc[j] = __builtin_amdgcn_wmma_f32_16x16x32_f16(
                false, a, false, b, (short)0, acc[j], false, false);
        }
    }
    const int mbase = (lane >> 4) << 3;
    const int nl    = lane & 15;
#pragma unroll
    for (int j = 0; j < 4; ++j) {
        const int n   = n0 + 16 * j + nl;
        const int qkv = n >> 10;            // 0:q 1:k 2:v
        const int rem = n & 1023;           // = d*16 + h (split order (3, D, H))
        const int d   = rem >> 4;
        const int h   = rem & 15;
        const float bn = bias[n];
#pragma unroll
        for (int r = 0; r < 8; ++r) {
            const int m  = m0 + mbase + r;
            const int s  = m & (SS - 1);
            const int bv = m >> 9;
            const int bh = bv * HH + h;
            const _Float16 val = (_Float16)(acc[j][r] + bn);
            if (qkv == 0)
                Qh[(size_t)bh * SS * DD + (size_t)s * DD + d] = val;
            else if (qkv == 1)
                Kh[(size_t)bh * SS * DD + (size_t)s * DD + d] = val;
            else
                Vt[(size_t)bh * DD * SS + (size_t)d * SS + s] = val;
        }
    }
}

// ---------------- Flash attention: one wave per (bh, 16-row q tile) ----------------
// K/V tiles are streamed into LDS by the Tensor Data Mover (double-buffered);
// fragments are then read with ds loads while the next tiles are in flight.
__global__ __launch_bounds__(32)
void attn_kernel(const _Float16* __restrict__ Qh,
                 const _Float16* __restrict__ Kh,
                 const _Float16* __restrict__ Vt,
                 _Float16* __restrict__ Oh)              // (16384, 1024) = (bv*s, h*64+d)
{
    __shared__ alignas(32) _Float16 Kl[2][32 * 64];      // K tile: (t, d), 4KB each
    __shared__ alignas(32) _Float16 Vl[2][64 * 32];      // V tile: (d, t), 4KB each
    __shared__ alignas(32) _Float16 Pl[16 * 32];         // P tile staging (C->A relayout)

    const int lane = threadIdx.x & 31;
    const int bh = blockIdx.x;                           // 0..511
    const int q0 = blockIdx.y * 16;                      // 0..496
    const _Float16* Qb = Qh + (size_t)bh * SS * DD;
    const _Float16* Kb = Kh + (size_t)bh * SS * DD;
    const _Float16* Vb = Vt + (size_t)bh * DD * SS;

    const unsigned koff0 = (unsigned)(size_t)&Kl[0][0];
    const unsigned koff1 = (unsigned)(size_t)&Kl[1][0];
    const unsigned voff0 = (unsigned)(size_t)&Vl[0][0];
    const unsigned voff1 = (unsigned)(size_t)&Vl[1][0];

    // prologue: TDM-fetch tiles for t0 = 0 into buffer 0
    tdm_load_2d(koff0, Kb, /*row_dw=*/32, /*nrows=*/32, /*stride_dw=*/32);
    tdm_load_2d(voff0, Vb, /*row_dw=*/16, /*nrows=*/64, /*stride_dw=*/256);

    const v16h aq0 = load_a16(Qb + (size_t)q0 * DD, DD);       // d = 0..31
    const v16h aq1 = load_a16(Qb + (size_t)q0 * DD + 32, DD);  // d = 32..63

    v8f o[4] = {};
    float mst[8], lst[8];
#pragma unroll
    for (int r = 0; r < 8; ++r) { mst[r] = -3.0e38f; lst[r] = 0.0f; }
    const int mbase = (lane >> 4) << 3;
    const int nl    = lane & 15;
    const int kb2   = (lane >> 4) << 4;                  // B-frag K base: 0 or 16
    const float scale = 0.125f;                          // 1/sqrt(64)

    for (int t0 = 0; t0 < SS; t0 += 32) {
        const int buf = (t0 >> 5) & 1;
        if (t0 + 32 < SS) {                              // prefetch next tiles
            const int nb = buf ^ 1;
            tdm_load_2d(nb ? koff1 : koff0, Kb + (size_t)(t0 + 32) * DD, 32, 32, 32);
            tdm_load_2d(nb ? voff1 : voff0, Vb + (t0 + 32), 16, 64, 256);
            __builtin_amdgcn_s_wait_tensorcnt((short)2); // current pair landed (in-order)
        } else {
            __builtin_amdgcn_s_wait_tensorcnt((short)0);
        }
        asm volatile("" ::: "memory");                   // LDS now holds TDM data

        // scores for two 16x16 t-tiles, K fragments from LDS (ds loads)
        v8f s0 = {}, s1 = {};
        {
            v16h b0 = *(const v16h*)&Kl[buf][(nl)      * 64 + kb2];        // sub0, d 0..31
            v16h b1 = *(const v16h*)&Kl[buf][(nl)      * 64 + 32 + kb2];   // sub0, d 32..63
            s0 = __builtin_amdgcn_wmma_f32_16x16x32_f16(false, aq0, false, b0, (short)0, s0, false, false);
            s0 = __builtin_amdgcn_wmma_f32_16x16x32_f16(false, aq1, false, b1, (short)0, s0, false, false);
            v16h b2 = *(const v16h*)&Kl[buf][(16 + nl) * 64 + kb2];        // sub1, d 0..31
            v16h b3 = *(const v16h*)&Kl[buf][(16 + nl) * 64 + 32 + kb2];   // sub1, d 32..63
            s1 = __builtin_amdgcn_wmma_f32_16x16x32_f16(false, aq0, false, b2, (short)0, s1, false, false);
            s1 = __builtin_amdgcn_wmma_f32_16x16x32_f16(false, aq1, false, b3, (short)0, s1, false, false);
        }
        // online softmax: row stats replicated across each 16-lane group
        float alpha[8];
#pragma unroll
        for (int r = 0; r < 8; ++r) {
            float a = s0[r] * scale, b = s1[r] * scale;
            float mx = fmaxf(a, b);
#pragma unroll
            for (int off = 1; off < 16; off <<= 1)
                mx = fmaxf(mx, __shfl_xor(mx, off, 16));
            const float mnew = fmaxf(mst[r], mx);
            alpha[r] = __expf(mst[r] - mnew);
            mst[r]   = mnew;
            const float p0 = __expf(a - mnew);
            const float p1 = __expf(b - mnew);
            s0[r] = p0; s1[r] = p1;
            float sum = p0 + p1;
#pragma unroll
            for (int off = 1; off < 16; off <<= 1)
                sum += __shfl_xor(sum, off, 16);
            lst[r] = lst[r] * alpha[r] + sum;
        }
#pragma unroll
        for (int j = 0; j < 4; ++j)
#pragma unroll
            for (int r = 0; r < 8; ++r)
                o[j][r] *= alpha[r];
        // C-layout -> LDS (16x32 P tile)
#pragma unroll
        for (int r = 0; r < 8; ++r) {
            Pl[(mbase + r) * 32 + nl]      = (_Float16)s0[r];
            Pl[(mbase + r) * 32 + 16 + nl] = (_Float16)s1[r];
        }
        __syncthreads();
        // LDS -> A-fragment of P
        v16h ap;
        {
            const int row = nl;
            const int kb  = (lane >> 4) << 4;
#pragma unroll
            for (int i = 0; i < 8; ++i) {
                ap[i]     = Pl[row * 32 + kb + i];
                ap[i + 8] = Pl[row * 32 + kb + 8 + i];
            }
        }
        __syncthreads();
        // O += P @ V, V fragments from LDS (tile stored (d, t))
#pragma unroll
        for (int j = 0; j < 4; ++j) {
            v16h bv = *(const v16h*)&Vl[buf][(16 * j + nl) * 32 + kb2];
            o[j] = __builtin_amdgcn_wmma_f32_16x16x32_f16(false, ap, false, bv, (short)0, o[j], false, false);
        }
    }
    // normalize and store: out channel = h*64 + d
    const int bvi = bh / HH, h = bh - bvi * HH;
#pragma unroll
    for (int j = 0; j < 4; ++j) {
        const int col = h * DD + 16 * j + nl;
#pragma unroll
        for (int r = 0; r < 8; ++r) {
            const int row = bvi * SS + q0 + mbase + r;
            Oh[(size_t)row * EE + col] = (_Float16)(o[j][r] / lst[r]);
        }
    }
}

// ---------------- output projection GEMM: Oh(16384x1024) @ Wo + bo -> fp32 ----------------
__global__ __launch_bounds__(256)
void out_gemm_kernel(const _Float16* __restrict__ Oh,
                     const _Float16* __restrict__ WT,    // (1024, 1024)
                     const float*    __restrict__ bias,  // (1024)
                     float* __restrict__ out)            // (16384, 1024)
{
    const int lane   = threadIdx.x & 31;
    const int wave   = blockIdx.x * (blockDim.x >> 5) + (threadIdx.x >> 5);
    const int ntiles = EE / 64;             // 16
    const int mt = wave / ntiles;
    const int nt = wave - mt * ntiles;
    if (mt >= MROWS / 16) return;
    const int m0 = mt * 16, n0 = nt * 64;

    v8f acc[4] = {};
    const _Float16* Arow = Oh + (size_t)m0 * EE;
    for (int k0 = 0; k0 < EE; k0 += 32) {
        v16h a = load_a16(Arow + k0, EE);
        __builtin_prefetch(Arow + k0 + 32, 0, 1);
#pragma unroll
        for (int j = 0; j < 4; ++j) {
            v16h b = load_bT16(WT + (size_t)(n0 + 16 * j) * EE + k0, EE);
            acc[j] = __builtin_amdgcn_wmma_f32_16x16x32_f16(
                false, a, false, b, (short)0, acc[j], false, false);
        }
    }
    const int mbase = (lane >> 4) << 3;
    const int nl    = lane & 15;
#pragma unroll
    for (int j = 0; j < 4; ++j) {
        const int n  = n0 + 16 * j + nl;
        const float bn = bias[n];
#pragma unroll
        for (int r = 0; r < 8; ++r) {
            const int m = m0 + mbase + r;
            out[(size_t)m * EE + n] = acc[j][r] + bn;
        }
    }
}

// ---------------- host-side launch ----------------
extern "C" void kernel_launch(void* const* d_in, const int* in_sizes, int n_in,
                              void* d_out, int out_size, void* d_ws, size_t ws_size,
                              hipStream_t stream) {
    const float* X    = (const float*)d_in[0];   // (4,8,512,1024)
    const float* Wqkv = (const float*)d_in[1];   // (1024,3072)
    const float* bqkv = (const float*)d_in[2];   // (3072)
    const float* Wo   = (const float*)d_in[3];   // (1024,1024)
    const float* bo   = (const float*)d_in[4];   // (1024)
    float* out = (float*)d_out;

    char* ws = (char*)d_ws;
    _Float16* Xh  = (_Float16*)ws; ws += (size_t)MROWS * EE * sizeof(_Float16);
    _Float16* WqT = (_Float16*)ws; ws += (size_t)NQKV * EE * sizeof(_Float16);
    _Float16* WoT = (_Float16*)ws; ws += (size_t)EE * EE * sizeof(_Float16);
    _Float16* Qh  = (_Float16*)ws; ws += (size_t)BHN * SS * DD * sizeof(_Float16);
    _Float16* Kh  = (_Float16*)ws; ws += (size_t)BHN * SS * DD * sizeof(_Float16);
    _Float16* Vt  = (_Float16*)ws; ws += (size_t)BHN * DD * SS * sizeof(_Float16);
    _Float16* Oh  = (_Float16*)ws; ws += (size_t)MROWS * EE * sizeof(_Float16);

    {   // fp32 -> f16 conversions / weight transposes
        int n = MROWS * EE;
        cvt_x_kernel<<<(n + 255) / 256, 256, 0, stream>>>(X, Xh, n);
        n = EE * NQKV;
        cvt_wT_kernel<<<(n + 255) / 256, 256, 0, stream>>>(Wqkv, WqT, EE, NQKV);
        n = EE * EE;
        cvt_wT_kernel<<<(n + 255) / 256, 256, 0, stream>>>(Wo, WoT, EE, EE);
    }
    {   // QKV projection: (16384/16)*(3072/64) = 49152 waves, 8 waves/block
        int waves = (MROWS / 16) * (NQKV / 64);
        qkv_gemm_kernel<<<waves / 8, 256, 0, stream>>>(Xh, WqT, bqkv, Qh, Kh, Vt);
    }
    {   // attention: 512 bh slices x 32 q-tiles, one wave each
        dim3 grid(BHN, SS / 16, 1);
        attn_kernel<<<grid, 32, 0, stream>>>(Qh, Kh, Vt, Oh);
    }
    {   // output projection: (16384/16)*(1024/64) = 16384 waves
        int waves = (MROWS / 16) * (EE / 64);
        out_gemm_kernel<<<waves / 8, 256, 0, stream>>>(Oh, WoT, bo, out);
    }
}
